// LinearAttention_7945689497705
// MI455X (gfx1250) — matrix operand order repaired and checked by reading, not compiled, not run on hardware
//
#include <hip/hip_runtime.h>
#include <hip/hip_bf16.h>

typedef __attribute__((ext_vector_type(8)))  _Float16 v8h;
typedef __attribute__((ext_vector_type(16))) _Float16 v16h;
typedef __attribute__((ext_vector_type(8)))  float    v8f;

#define BATCH 16
#define CIN   256
#define HW    4096
#define HEADS 4
#define DH    32
#define HID   128
#define QKV_M 384
#define NMEM  4

static __device__ __forceinline__ v16h cat16(v8h lo, v8h hi) {
    return __builtin_shufflevector(lo, hi, 0,1,2,3,4,5,6,7,8,9,10,11,12,13,14,15);
}
static __device__ __forceinline__ v8f wmma_f16(v16h a, v16h b, v8f c) {
    return __builtin_amdgcn_wmma_f32_16x16x32_f16(false, a, false, b, (short)0, c, false, false);
}

// ---------------------------------------------------------------------------
// K0: fold (gamma1+1) into w_qkv columns; convert weights to f16
// ---------------------------------------------------------------------------
__global__ __launch_bounds__(256) void k0_prep(const float* __restrict__ w_qkv,
                                               const float* __restrict__ gamma1,
                                               const float* __restrict__ w_out,
                                               _Float16* __restrict__ wq16,
                                               _Float16* __restrict__ wo16) {
    int idx = blockIdx.x * 256 + threadIdx.x;
    if (idx < QKV_M * CIN) {
        int c = idx & (CIN - 1);
        wq16[idx] = (_Float16)(w_qkv[idx] * (gamma1[c] + 1.0f));
    }
    if (idx < CIN * HID) wo16[idx] = (_Float16)w_out[idx];
}

// ---------------------------------------------------------------------------
// K1: fused RMSNorm + qkv projection.  GEMM: [384 x 256] x [256 x 16 pixels]
// 8 waves, each wave computes three 16x16 D tiles (full M=384 per workgroup).
// ---------------------------------------------------------------------------
__global__ __launch_bounds__(256) void k1_qkv(const float* __restrict__ x,
                                              const _Float16* __restrict__ wq,
                                              _Float16* __restrict__ qkv) {
    __shared__ _Float16 bt[16][CIN];    // [pixel][channel]  (K contiguous per pixel)
    __shared__ float    psum[16][17];
    __shared__ float    pscale[16];

    int t   = threadIdx.x;
    int blk = blockIdx.x;               // b * 256 + ntile
    int b   = blk >> 8;
    int n0  = (blk & 255) * 16;

    // ---- stage: per-pixel sumsq + normalized f16 activations --------------
    int p = t & 15, g = t >> 4;         // pixel, 16-channel group
    const float* xb = x + (size_t)b * CIN * HW + n0 + p;
    float xv[16];
    float ss = 0.f;
#pragma unroll
    for (int i = 0; i < 16; ++i) {
        float v = xb[(size_t)(g * 16 + i) * HW];
        xv[i] = v; ss += v * v;
    }
    psum[p][g] = ss;
    __syncthreads();
    if (t < 16) {
        float s = 0.f;
#pragma unroll
        for (int i = 0; i < 16; ++i) s += psum[t][i];
        pscale[t] = 16.0f / fmaxf(sqrtf(s), 1e-12f);   // sqrt(C)=16 folded here
    }
    __syncthreads();
    float sc = pscale[p];
#pragma unroll
    for (int i = 0; i < 16; ++i) bt[p][g * 16 + i] = (_Float16)(xv[i] * sc);
    __syncthreads();

    // ---- WMMA GEMM --------------------------------------------------------
    int wv = t >> 5, lane = t & 31;
    int l15 = lane & 15, kh = lane >> 4;
    v8f acc[3] = {};
#pragma unroll
    for (int s = 0; s < 8; ++s) {                       // K loop: 8 x 32
        int k0 = s * 32;
        const _Float16* bp = &bt[l15][k0 + kh * 8];     // B: N=l15, K=8*kh+e (e<8), 16+8*kh+(e-8)
        v16h bf = cat16(*(const v8h*)bp, *(const v8h*)(bp + 16));
#pragma unroll
        for (int i = 0; i < 3; ++i) {
            int m0 = (wv * 3 + i) * 16;
            const _Float16* ap = wq + (size_t)(m0 + l15) * CIN + k0 + kh * 8;
            v16h af = cat16(*(const v8h*)ap, *(const v8h*)(ap + 16));
            acc[i] = wmma_f16(af, bf, acc[i]);
        }
    }
    // ---- store D: row m0 + j + 8*kh, col = l15 ----------------------------
#pragma unroll
    for (int i = 0; i < 3; ++i) {
        int m0 = (wv * 3 + i) * 16;
        _Float16* op = qkv + ((size_t)b * QKV_M + m0 + kh * 8) * HW + n0 + l15;
#pragma unroll
        for (int j = 0; j < 8; ++j) op[(size_t)j * HW] = (_Float16)acc[i][j];
    }
}

// ---------------------------------------------------------------------------
// K2a: k-softmax row stats (max, 1/sumexp) over n = 4 mem + 4096 pixels.
// One wave per (b,h,d) row.
// ---------------------------------------------------------------------------
__global__ __launch_bounds__(256) void k2a_kstats(const _Float16* __restrict__ qkv,
                                                  const float* __restrict__ mem_kv,
                                                  float* __restrict__ rowmax,
                                                  float* __restrict__ rowinv) {
    int wv = threadIdx.x >> 5, lane = threadIdx.x & 31;
    int r = blockIdx.x * 8 + wv;                 // 0..2047
    int b = r >> 7, rem = r & 127, h = rem >> 5, d = rem & 31;
    const _Float16* kp = qkv + ((size_t)b * QKV_M + HID + rem) * HW;
    const float*    mk = mem_kv + ((size_t)h * DH + d) * NMEM;   // mem_kv[0]

    float m = -1e30f;
    for (int n = lane; n < HW; n += 32) m = fmaxf(m, (float)kp[n]);
    if (lane < NMEM) m = fmaxf(m, mk[lane]);
#pragma unroll
    for (int off = 16; off; off >>= 1) m = fmaxf(m, __shfl_xor(m, off, 32));

    float s = 0.f;
    for (int n = lane; n < HW; n += 32) s += __expf((float)kp[n] - m);
    if (lane < NMEM) s += __expf(mk[lane] - m);
#pragma unroll
    for (int off = 16; off; off >>= 1) s += __shfl_xor(s, off, 32);

    if (lane == 0) { rowmax[r] = m; rowinv[r] = 1.0f / s; }
}

// ---------------------------------------------------------------------------
// K2b: context[b,h,d,e] = sum_n softmax_n(k)[d,n] * v[e,n]   (n = 4+4096)
// One block per (b,h); LDS-tiled 32-wide chunks.
// ---------------------------------------------------------------------------
__global__ __launch_bounds__(256) void k2b_context(const _Float16* __restrict__ qkv,
                                                   const float* __restrict__ mem_kv,
                                                   const float* __restrict__ rowmax,
                                                   const float* __restrict__ rowinv,
                                                   float* __restrict__ ctx) {
    __shared__ float kex[32][33];
    __shared__ float vvs[32][33];
    int t = threadIdx.x;
    int bh = blockIdx.x;                 // 0..63
    int b = bh >> 2, h = bh & 3;
    int rbase = b * 128 + h * 32;
    int dl = t >> 3, nl = t & 7;         // loader row, loader col base
    float rm = rowmax[rbase + dl], ri = rowinv[rbase + dl];

    const _Float16* kp = qkv + ((size_t)b * QKV_M + HID       + h * DH + dl) * HW;
    const _Float16* vp = qkv + ((size_t)b * QKV_M + 2 * HID   + h * DH + dl) * HW;
    const float*    mk = mem_kv + ((size_t)h * DH + dl) * NMEM;
    const float*    mv = mem_kv + ((size_t)(HEADS + h) * DH + dl) * NMEM;

    int eb = (t & 7) * 4;
    float acc[4] = {0.f, 0.f, 0.f, 0.f};

    for (int c0 = 0; c0 < NMEM + HW; c0 += 32) {
#pragma unroll
        for (int q = 0; q < 4; ++q) {
            int nn = nl + 8 * q;
            int n  = c0 + nn;
            float kv = 0.f, vv = 0.f;
            if (n < NMEM)            { kv = __expf(mk[n] - rm) * ri;            vv = mv[n]; }
            else if (n < NMEM + HW)  { int j = n - NMEM;
                                       kv = __expf((float)kp[j] - rm) * ri;     vv = (float)vp[j]; }
            kex[dl][nn] = kv; vvs[dl][nn] = vv;
        }
        __syncthreads();
#pragma unroll 4
        for (int nn = 0; nn < 32; ++nn) {
            float kd = kex[dl][nn];
#pragma unroll
            for (int j = 0; j < 4; ++j) acc[j] += kd * vvs[eb + j][nn];
        }
        __syncthreads();
    }
    float* cp = ctx + ((size_t)bh * DH + dl) * DH + eb;
#pragma unroll
    for (int j = 0; j < 4; ++j) cp[j] = acc[j];
}

// ---------------------------------------------------------------------------
// K3: q-softmax (over d) + apply context:  attn[e,n] = sum_d ctx[d,e]*qs[d,n]
// One block = one (b,h) x 256-pixel chunk; one thread = one pixel.
// ---------------------------------------------------------------------------
__global__ __launch_bounds__(256) void k3_apply(const _Float16* __restrict__ qkv,
                                                const float* __restrict__ ctx,
                                                _Float16* __restrict__ attn) {
    __shared__ float ctxs[32][32];
    __shared__ float qs[32][257];
    int t = threadIdx.x;
    int blk = blockIdx.x;                 // bh*16 + chunk
    int bh = blk >> 4, n0 = (blk & 15) * 256;
    int b = bh >> 2, h = bh & 3;

    const float* cp = ctx + (size_t)bh * 1024;
#pragma unroll
    for (int i = 0; i < 4; ++i) ((float*)ctxs)[t * 4 + i] = cp[t * 4 + i];

    const _Float16* qp = qkv + ((size_t)b * QKV_M + h * DH) * HW + n0;
#pragma unroll 4
    for (int r = 0; r < 32; ++r) qs[r][t] = (float)qp[(size_t)r * HW + t];
    __syncthreads();

    float qv[32];
    float m = -1e30f;
#pragma unroll
    for (int r = 0; r < 32; ++r) { qv[r] = qs[r][t]; m = fmaxf(m, qv[r]); }
    float s = 0.f;
#pragma unroll
    for (int r = 0; r < 32; ++r) { qv[r] = __expf(qv[r] - m); s += qv[r]; }
    float sc = 0.17677669529663687f / s;          // DH^-0.5 / sumexp

    float acc[32];
#pragma unroll
    for (int e = 0; e < 32; ++e) acc[e] = 0.f;
    for (int d = 0; d < 32; ++d) {
        float qd = qv[d] * sc;
#pragma unroll
        for (int e = 0; e < 32; ++e) acc[e] += ctxs[d][e] * qd;
    }
    _Float16* op = attn + ((size_t)b * HID + h * DH) * HW + n0 + t;
#pragma unroll
    for (int e = 0; e < 32; ++e) op[(size_t)e * HW] = (_Float16)acc[e];
}

// ---------------------------------------------------------------------------
// K4: fused output projection + bias + RMSNorm.  GEMM [256 x 128] x [128 x 16]
// full M per workgroup so the per-pixel column norm can be fused (LDS f32 atomics).
// ---------------------------------------------------------------------------
__global__ __launch_bounds__(256) void k4_out(const _Float16* __restrict__ attn,
                                              const _Float16* __restrict__ wo,
                                              const float* __restrict__ b_out,
                                              const float* __restrict__ gamma2,
                                              float* __restrict__ out) {
    __shared__ _Float16 bt[16][HID];    // [pixel][channel]
    __shared__ float    colsq[16];
    int t = threadIdx.x;
    int blk = blockIdx.x;
    int b = blk >> 8, n0 = (blk & 255) * 16;

    {   // stage B tile (transpose to pixel-major)
        int col = t & 15, r0 = t >> 4;
#pragma unroll
        for (int it = 0; it < 8; ++it) {
            int row = r0 + it * 16;
            bt[col][row] = attn[((size_t)b * HID + row) * HW + n0 + col];
        }
    }
    if (t < 16) colsq[t] = 0.f;
    __syncthreads();

    int wv = t >> 5, lane = t & 31;
    int l15 = lane & 15, kh = lane >> 4;
    v8f acc[2] = {};
#pragma unroll
    for (int s = 0; s < 4; ++s) {                       // K loop: 4 x 32
        int k0 = s * 32;
        const _Float16* bp = &bt[l15][k0 + kh * 8];
        v16h bf = cat16(*(const v8h*)bp, *(const v8h*)(bp + 16));
#pragma unroll
        for (int i = 0; i < 2; ++i) {
            int m0 = (wv * 2 + i) * 16;
            const _Float16* ap = wo + (size_t)(m0 + l15) * HID + k0 + kh * 8;
            v16h af = cat16(*(const v8h*)ap, *(const v8h*)(ap + 16));
            acc[i] = wmma_f16(af, bf, acc[i]);
        }
    }
    // bias + per-column sum of squares
    float vals[2][8];
#pragma unroll
    for (int i = 0; i < 2; ++i) {
        int m0 = (wv * 2 + i) * 16;
        float part = 0.f;
#pragma unroll
        for (int j = 0; j < 8; ++j) {
            int m = m0 + kh * 8 + j;
            float v = acc[i][j] + b_out[m];
            vals[i][j] = v; part += v * v;
        }
        atomicAdd(&colsq[l15], part);
    }
    __syncthreads();
    float inv = 16.0f / fmaxf(sqrtf(colsq[l15]), 1e-12f);
#pragma unroll
    for (int i = 0; i < 2; ++i) {
        int m0 = (wv * 2 + i) * 16;
#pragma unroll
        for (int j = 0; j < 8; ++j) {
            int m = m0 + kh * 8 + j;
            out[((size_t)b * CIN + m) * HW + n0 + l15] =
                vals[i][j] * (gamma2[m] + 1.0f) * inv;
        }
    }
}

// ---------------------------------------------------------------------------
extern "C" void kernel_launch(void* const* d_in, const int* in_sizes, int n_in,
                              void* d_out, int out_size, void* d_ws, size_t ws_size,
                              hipStream_t stream) {
    const float* x      = (const float*)d_in[0];
    const float* gamma1 = (const float*)d_in[1];
    const float* mem_kv = (const float*)d_in[2];
    const float* w_qkv  = (const float*)d_in[3];
    const float* w_out  = (const float*)d_in[4];
    const float* b_out  = (const float*)d_in[5];
    const float* gamma2 = (const float*)d_in[6];
    float* out = (float*)d_out;

    char* ws = (char*)d_ws;
    size_t off = 0;
    auto take = [&](size_t bytes) { size_t r = off; off = (off + bytes + 255) & ~(size_t)255; return r; };
    _Float16* wq16   = (_Float16*)(ws + take((size_t)QKV_M * CIN * 2));
    _Float16* wo16   = (_Float16*)(ws + take((size_t)CIN * HID * 2));
    _Float16* qkv    = (_Float16*)(ws + take((size_t)BATCH * QKV_M * HW * 2));
    float*    rowmax = (float*)   (ws + take((size_t)BATCH * 128 * 4));
    float*    rowinv = (float*)   (ws + take((size_t)BATCH * 128 * 4));
    float*    ctx    = (float*)   (ws + take((size_t)BATCH * HEADS * DH * DH * 4));
    _Float16* attn   = (_Float16*)(ws + take((size_t)BATCH * HID * HW * 2));
    (void)in_sizes; (void)n_in; (void)out_size; (void)ws_size;

    k0_prep<<<(QKV_M * CIN + 255) / 256, 256, 0, stream>>>(w_qkv, gamma1, w_out, wq16, wo16);
    k1_qkv<<<BATCH * (HW / 16), 256, 0, stream>>>(x, wq16, qkv);
    k2a_kstats<<<(BATCH * 128) / 8, 256, 0, stream>>>(qkv, mem_kv, rowmax, rowinv);
    k2b_context<<<BATCH * HEADS, 256, 0, stream>>>(qkv, mem_kv, rowmax, rowinv, ctx);
    k3_apply<<<BATCH * HEADS * (HW / 256), 256, 0, stream>>>(qkv, ctx, attn);
    k4_out<<<BATCH * (HW / 16), 256, 0, stream>>>(attn, wo16, b_out, gamma2, out);
}
